// GAT_23880018166267
// MI455X (gfx1250) — matrix operand order repaired and checked by reading, not compiled, not run on hardware
//
#include <hip/hip_runtime.h>
#include <hip/hip_fp16.h>
#include <stdint.h>
#include <math.h>

// ---------------------------------------------------------------------------
// GAT (2-layer, PyG GATConv semantics) for MI455X / gfx1250.
// GEMMs: v_wmma_f32_16x16x32_f16, fragment-packed operands (b128 loads only).
// Edge softmax / aggregation: fp32, float4 scores, atomics, no div/mod.
// ---------------------------------------------------------------------------

typedef __attribute__((ext_vector_type(16))) _Float16 v16h;
typedef __attribute__((ext_vector_type(8)))  float    v8f;

#define NEG_SLOPE 0.2f
#define NHEAD 4            // model constant (H)
#define F_IN  128          // model constant (input feature width)

__device__ __forceinline__ void atomicMaxF(float* addr, float val) {
  // monotone float-bits trick: positives via signed max, negatives via unsigned min
  if (val >= 0.0f) atomicMax((int*)addr, __float_as_int(val));
  else             atomicMin((unsigned int*)addr, __float_as_uint(val));
}

__device__ __forceinline__ float lrelu(float v) {
  return (v > 0.f) ? v : NEG_SLOPE * v;
}

// ---------------- column mean / inv-std (ddof=1) ---------------------------
__global__ void __launch_bounds__(256)
k_colstats(const float* __restrict__ x, int Nrows,
           float* __restrict__ mean, float* __restrict__ istd) {
  const int c = blockIdx.x;            // one block per feature column
  __shared__ float ssum[256], ssq[256];
  float s = 0.f, q = 0.f;
  for (int r = threadIdx.x; r < Nrows; r += 256) {
    float v = x[(size_t)r * F_IN + c];
    s += v; q += v * v;
  }
  ssum[threadIdx.x] = s; ssq[threadIdx.x] = q;
  __syncthreads();
  for (int o = 128; o > 0; o >>= 1) {
    if (threadIdx.x < o) {
      ssum[threadIdx.x] += ssum[threadIdx.x + o];
      ssq[threadIdx.x]  += ssq[threadIdx.x + o];
    }
    __syncthreads();
  }
  if (threadIdx.x == 0) {
    float mu  = ssum[0] / (float)Nrows;
    float var = (ssq[0] - (float)Nrows * mu * mu) / (float)(Nrows - 1);
    mean[c] = mu;
    istd[c] = rsqrtf(var);
  }
}

// ---------------- standardize + cast to f16 --------------------------------
__global__ void __launch_bounds__(256)
k_standardize_half(const float* __restrict__ x, const float* __restrict__ mean,
                   const float* __restrict__ istd, _Float16* __restrict__ xh,
                   long long total) {
  long long i = (long long)blockIdx.x * blockDim.x + threadIdx.x;
  if (i >= total) return;
  int c = (int)(i & (F_IN - 1));
  xh[i] = (_Float16)((x[i] - mean[c]) * istd[c]);
}

// ---------------- pack W (f32 row-major K x Ncol) into WMMA B fragments ----
// Packed layout: frag f = ktile*ntiles + ntile; element ((f*32 + lane)*16 + i)
//   = W[(ktile*32 + (lane>>4)*16 + i) * Ncol + ntile*16 + (lane&15)]  as f16.
// Lane's 16 halves are contiguous (32B) -> two global_load_b128 in the GEMM.
__global__ void __launch_bounds__(256)
k_pack_b(const float* __restrict__ W, _Float16* __restrict__ Bp, int K, int Ncol) {
  int idx = blockIdx.x * blockDim.x + threadIdx.x;
  int total = K * Ncol;
  if (idx >= total) return;
  int i    = idx & 15;
  int lane = (idx >> 4) & 31;
  int f    = idx >> 9;
  int ntiles = Ncol >> 4;
  int ktile = f / ntiles, ntile = f - ktile * ntiles;
  int k = ktile * 32 + (lane >> 4) * 16 + i;
  int n = ntile * 16 + (lane & 15);
  Bp[idx] = (_Float16)W[(size_t)k * Ncol + n];
}

// ---------------- WMMA GEMM: C[M x Ncol] = A[M x K] * Bp ------------------
// A row-major f16; Bp fragment-packed f16; C f32.
// Block = 256 threads = 8 waves, one 16-row stripe; wave w owns N-tiles
// {w, w+8}. A tile staged in LDS *fragment-ordered* so each lane reads its
// v16h with two ds_load_b128.
__global__ void __launch_bounds__(256)
k_wmma_gemm(const _Float16* __restrict__ A, const _Float16* __restrict__ Bp,
            float* __restrict__ C, int M, int K, int Ncol) {
  __shared__ __align__(16) uint32_t As32[32 * 8];   // 32 lanes x 8 dwords (1 KB)
  const int lane   = threadIdx.x & 31;
  const int wave   = threadIdx.x >> 5;              // 0..7
  const int ntiles = Ncol >> 4;
  const int row0   = blockIdx.x * 16;

  const int  t0   = wave;                           // ntiles >= 10 > 7 always
  const int  t1   = wave + 8;
  const bool has1 = (t1 < ntiles);

  v8f acc0, acc1;
  #pragma unroll
  for (int r = 0; r < 8; ++r) { acc0[r] = 0.f; acc1[r] = 0.f; }

  // cooperative-loader coordinates: thread handles one dword (2 halves)
  const int lr = threadIdx.x >> 4;                  // tile row 0..15
  const int lc = (threadIdx.x & 15) * 2;            // tile col 0..30 (even)
  int lrr = row0 + lr; if (lrr >= M) lrr = M - 1;   // clamp (M exact anyway)
  const uint32_t* __restrict__ Arow =
      (const uint32_t*)(A + (size_t)lrr * K);       // K even, 4B aligned
  // inverse fragment mapping for LDS write (both halves land in same lane)
  const int w_lane = lr + ((lc & 16) ? 16 : 0);
  const int w_dw   = (lc & 15) >> 1;

  const int nk = K >> 5;
  for (int kt = 0; kt < nk; ++kt) {
    const int k0 = kt << 5;
    As32[w_lane * 8 + w_dw] = Arow[(k0 + lc) >> 1];
    if (kt + 1 < nk)                                 // hide next tile's latency
      __builtin_prefetch(Arow + ((k0 + 32 + lc) >> 1), 0, 1);
    __syncthreads();

    const v16h af = *(const v16h*)&As32[lane * 8];   // 2x ds_load_b128

    {
      const v16h bf = *(const v16h*)(Bp + (((size_t)(kt * ntiles + t0) * 32 + lane) << 4));
      acc0 = __builtin_amdgcn_wmma_f32_16x16x32_f16(false, af, false, bf,
                                                    (short)0, acc0, false, false);
    }
    if (has1) {
      const v16h bf = *(const v16h*)(Bp + (((size_t)(kt * ntiles + t1) * 32 + lane) << 4));
      acc1 = __builtin_amdgcn_wmma_f32_16x16x32_f16(false, af, false, bf,
                                                    (short)0, acc1, false, false);
    }
    __syncthreads();
  }

  // C/D layout: VGPR r -> M = r (lanes 0-15) / 8+r (lanes 16-31), N = lane%16
  const int rbase = row0 + (lane >> 4) * 8;
  const int ncol0 = t0 * 16 + (lane & 15);
  #pragma unroll
  for (int r = 0; r < 8; ++r) {
    int rr = rbase + r;
    if (rr < M) C[(size_t)rr * Ncol + ncol0] = acc0[r];
  }
  if (has1) {
    const int ncol1 = t1 * 16 + (lane & 15);
    #pragma unroll
    for (int r = 0; r < 8; ++r) {
      int rr = rbase + r;
      if (rr < M) C[(size_t)rr * Ncol + ncol1] = acc1[r];
    }
  }
}

// ---------------- attention scores: a[n,h] = <h[n,h,:], att[h,:]> ----------
__global__ void __launch_bounds__(256)
k_att_scores(const float* __restrict__ h, const float* __restrict__ att,
             float* __restrict__ a, int Nnodes, int Cc) {
  int i = blockIdx.x * blockDim.x + threadIdx.x;   // over N*NHEAD
  if (i >= Nnodes * NHEAD) return;
  int hh = i & (NHEAD - 1), n = i >> 2;
  const float4* hp = (const float4*)(h + (size_t)n * NHEAD * Cc + (size_t)hh * Cc);
  const float4* ap = (const float4*)(att + (size_t)hh * Cc);
  float s = 0.f;
  const int c4 = Cc >> 2;
  for (int c = 0; c < c4; ++c) {
    float4 u = hp[c], v = ap[c];
    s += u.x * v.x + u.y * v.y + u.z * v.z + u.w * v.w;
  }
  a[i] = s;
}

// ---------------- helpers ---------------------------------------------------
__global__ void __launch_bounds__(256)
k_fill(float* __restrict__ p, float v, long long n) {
  long long i = (long long)blockIdx.x * blockDim.x + threadIdx.x;
  if (i < n) p[i] = v;
}

// blockIdx = node, threadIdx = feature (blockDim == HC)
__global__ void k_init_bias(float* __restrict__ out, const float* __restrict__ bias) {
  out[(size_t)blockIdx.x * blockDim.x + threadIdx.x] = bias[threadIdx.x];
}

// ---------------- edge pass 1: e = lrelu(a_src[s]+a_dst[d]); seg-max -------
// one thread per message, all NHEAD heads via float4
__global__ void __launch_bounds__(256)
k_edge_lrelu_max(const long long* __restrict__ ei, long long E, int Nnodes,
                 const float4* __restrict__ as, const float4* __restrict__ ad,
                 float4* __restrict__ e, float* __restrict__ mbuf) {
  long long eid = (long long)blockIdx.x * blockDim.x + threadIdx.x;
  if (eid >= E + Nnodes) return;
  long long s, d;
  if (eid < E) { s = ei[eid]; d = ei[E + eid]; } else { s = d = eid - E; }
  float4 u = as[s], v = ad[d];
  float4 r;
  r.x = lrelu(u.x + v.x);
  r.y = lrelu(u.y + v.y);
  r.z = lrelu(u.z + v.z);
  r.w = lrelu(u.w + v.w);
  e[eid] = r;
  float* mb = mbuf + d * NHEAD;
  atomicMaxF(mb + 0, r.x);
  atomicMaxF(mb + 1, r.y);
  atomicMaxF(mb + 2, r.z);
  atomicMaxF(mb + 3, r.w);
}

// ---------------- edge pass 2: e = exp(e - m[d]); seg-sum denom ------------
__global__ void __launch_bounds__(256)
k_edge_expsum(const long long* __restrict__ ei, long long E, int Nnodes,
              float4* __restrict__ e, const float4* __restrict__ mbuf,
              float* __restrict__ den) {
  long long eid = (long long)blockIdx.x * blockDim.x + threadIdx.x;
  if (eid >= E + Nnodes) return;
  long long d;
  if (eid < E) d = ei[E + eid]; else d = eid - E;
  float4 ev = e[eid], mv = mbuf[d];
  float4 r;
  r.x = expf(ev.x - mv.x);
  r.y = expf(ev.y - mv.y);
  r.z = expf(ev.z - mv.z);
  r.w = expf(ev.w - mv.w);
  e[eid] = r;
  float* dn = den + d * NHEAD;
  atomicAdd(dn + 0, r.x);
  atomicAdd(dn + 1, r.y);
  atomicAdd(dn + 2, r.z);
  atomicAdd(dn + 3, r.w);
}

// ---------------- edge pass 3: out[d] += h[s] * (e/den[d]) -----------------
// blockIdx = message, threadIdx = feature (blockDim == HC = NHEAD*Cc)
__global__ void k_edge_aggregate(const long long* __restrict__ ei, long long E,
                                 int Cc, const float* __restrict__ hfeat,
                                 const float* __restrict__ eexp,
                                 const float* __restrict__ den,
                                 float* __restrict__ out) {
  const long long eid = blockIdx.x;
  const int f  = threadIdx.x;
  const int HC = blockDim.x;
  const int hh = f / Cc;                    // uniform small division
  long long s, d;
  if (eid < E) { s = ei[eid]; d = ei[E + eid]; } else { s = d = eid - E; }
  float alpha = eexp[eid * NHEAD + hh] / den[d * NHEAD + hh];
  atomicAdd(&out[d * (long long)HC + f], hfeat[s * (long long)HC + f] * alpha);
}

// ---------------- relu + cast f16 (inter-layer activation) -----------------
__global__ void __launch_bounds__(256)
k_relu_tohalf(const float* __restrict__ src, _Float16* __restrict__ dst, long long n) {
  long long i = (long long)blockIdx.x * blockDim.x + threadIdx.x;
  if (i < n) dst[i] = (_Float16)fmaxf(src[i], 0.f);
}

// ===========================================================================
extern "C" void kernel_launch(void* const* d_in, const int* in_sizes, int n_in,
                              void* d_out, int out_size, void* d_ws, size_t ws_size,
                              hipStream_t stream) {
  const float*     x    = (const float*)d_in[0];
  const long long* ei   = (const long long*)d_in[1];   // int64 [2,E] row-major
  const float*     W1   = (const float*)d_in[2];
  const float*     as1  = (const float*)d_in[3];
  const float*     ad1  = (const float*)d_in[4];
  const float*     b1   = (const float*)d_in[5];
  const float*     W2   = (const float*)d_in[6];
  const float*     as2  = (const float*)d_in[7];
  const float*     ad2  = (const float*)d_in[8];
  const float*     b2   = (const float*)d_in[9];
  float*           out  = (float*)d_out;

  const int N    = in_sizes[0] / F_IN;             // 50000
  const long long E = (long long)in_sizes[1] / 2;  // 800000
  const int HID  = in_sizes[3] / NHEAD;            // 64
  const int CLS  = in_sizes[7] / NHEAD;            // 40
  const int HC1  = NHEAD * HID;                    // 256
  const int HC2  = NHEAD * CLS;                    // 160
  const long long TE = E + N;                      // messages incl. self loops

  size_t off = 0;
  auto alloc = [&](size_t bytes) -> void* {
    off = (off + 255) & ~(size_t)255;
    void* p = (char*)d_ws + off;
    off += bytes;
    return p;
  };
  float*     mean = (float*)alloc(sizeof(float) * F_IN);
  float*     istd = (float*)alloc(sizeof(float) * F_IN);
  _Float16*  xh   = (_Float16*)alloc(sizeof(_Float16) * (size_t)N * F_IN);
  _Float16*  b1p  = (_Float16*)alloc(sizeof(_Float16) * (size_t)F_IN * HC1);
  _Float16*  b2p  = (_Float16*)alloc(sizeof(_Float16) * (size_t)HC1 * HC2);
  float*     h1   = (float*)alloc(sizeof(float) * (size_t)N * HC1);
  float*     a1s  = (float*)alloc(sizeof(float) * (size_t)N * NHEAD);
  float*     a1d  = (float*)alloc(sizeof(float) * (size_t)N * NHEAD);
  float*     e1   = (float*)alloc(sizeof(float) * (size_t)TE * NHEAD);
  float*     m1   = (float*)alloc(sizeof(float) * (size_t)N * NHEAD);
  float*     dn1  = (float*)alloc(sizeof(float) * (size_t)N * NHEAD);
  float*     agg1 = (float*)alloc(sizeof(float) * (size_t)N * HC1);
  _Float16*  x2h  = (_Float16*)alloc(sizeof(_Float16) * (size_t)N * HC1);
  float*     h2   = (float*)alloc(sizeof(float) * (size_t)N * HC2);
  float*     a2s  = (float*)alloc(sizeof(float) * (size_t)N * NHEAD);
  float*     a2d  = (float*)alloc(sizeof(float) * (size_t)N * NHEAD);
  float*     e2   = (float*)alloc(sizeof(float) * (size_t)TE * NHEAD);
  float*     m2   = (float*)alloc(sizeof(float) * (size_t)N * NHEAD);
  float*     dn2  = (float*)alloc(sizeof(float) * (size_t)N * NHEAD);
  (void)ws_size;

  auto nb = [](long long total) { return (unsigned)((total + 255) / 256); };

  // ================= Layer 1 =================
  k_colstats<<<F_IN, 256, 0, stream>>>(x, N, mean, istd);
  k_standardize_half<<<nb((long long)N * F_IN), 256, 0, stream>>>(
      x, mean, istd, xh, (long long)N * F_IN);
  k_pack_b<<<nb((long long)F_IN * HC1), 256, 0, stream>>>(W1, b1p, F_IN, HC1);
  k_pack_b<<<nb((long long)HC1 * HC2), 256, 0, stream>>>(W2, b2p, HC1, HC2);

  k_wmma_gemm<<<(N + 15) / 16, 256, 0, stream>>>(xh, b1p, h1, N, F_IN, HC1);

  k_att_scores<<<nb((long long)N * NHEAD), 256, 0, stream>>>(h1, as1, a1s, N, HID);
  k_att_scores<<<nb((long long)N * NHEAD), 256, 0, stream>>>(h1, ad1, a1d, N, HID);

  k_fill<<<nb((long long)N * NHEAD), 256, 0, stream>>>(m1, -INFINITY, (long long)N * NHEAD);
  k_fill<<<nb((long long)N * NHEAD), 256, 0, stream>>>(dn1, 0.f, (long long)N * NHEAD);

  k_edge_lrelu_max<<<nb(TE), 256, 0, stream>>>(ei, E, N, (const float4*)a1s,
                                               (const float4*)a1d, (float4*)e1, m1);
  k_edge_expsum<<<nb(TE), 256, 0, stream>>>(ei, E, N, (float4*)e1,
                                            (const float4*)m1, dn1);

  k_init_bias<<<N, HC1, 0, stream>>>(agg1, b1);
  k_edge_aggregate<<<(unsigned)TE, HC1, 0, stream>>>(ei, E, HID, h1, e1, dn1, agg1);

  k_relu_tohalf<<<nb((long long)N * HC1), 256, 0, stream>>>(agg1, x2h, (long long)N * HC1);

  // ================= Layer 2 =================
  k_wmma_gemm<<<(N + 15) / 16, 256, 0, stream>>>(x2h, b2p, h2, N, HC1, HC2);

  k_att_scores<<<nb((long long)N * NHEAD), 256, 0, stream>>>(h2, as2, a2s, N, CLS);
  k_att_scores<<<nb((long long)N * NHEAD), 256, 0, stream>>>(h2, ad2, a2d, N, CLS);

  k_fill<<<nb((long long)N * NHEAD), 256, 0, stream>>>(m2, -INFINITY, (long long)N * NHEAD);
  k_fill<<<nb((long long)N * NHEAD), 256, 0, stream>>>(dn2, 0.f, (long long)N * NHEAD);

  k_edge_lrelu_max<<<nb(TE), 256, 0, stream>>>(ei, E, N, (const float4*)a2s,
                                               (const float4*)a2d, (float4*)e2, m2);
  k_edge_expsum<<<nb(TE), 256, 0, stream>>>(ei, E, N, (float4*)e2,
                                            (const float4*)m2, dn2);

  k_init_bias<<<N, HC2, 0, stream>>>(out, b2);
  k_edge_aggregate<<<(unsigned)TE, HC2, 0, stream>>>(ei, E, CLS, h2, e2, dn2, out);
}
